// FourierFeatureMLP_24223615549934
// MI455X (gfx1250) — matrix operand order repaired and checked by reading, not compiled, not run on hardware
//
#include <hip/hip_runtime.h>
#include <hip/hip_bf16.h>

#define NFREQ   12
#define EMB     75
#define EMBPAD  96
#define SA      264     // activation LDS row stride (halfwords): 528B, 16B aligned, 4-bank row stagger
#define SW      264     // weight LDS row stride (halfwords)
#define ROWS    128     // points per block
#define THREADS 256     // 8 waves (wave32)

// Workspace layout (halfword offsets): bf16 weights, transposed W^T[n][k], row stride = Kpad
#define WS_WDIN   0u            // 256 x 96
#define WS_HID    24576u        // 6 x (256 x 256)
#define WS_WDOUT  417792u       // 288 x 256
#define WS_WCIN   491520u       // 128 x 256
#define WS_WCHID  524288u       // 2 x (128 x 128)
#define WS_WCOUT  557056u       // 16 x 128
#define WS_TOTAL_HW 559104u
#define WS_TOTAL_BYTES (WS_TOTAL_HW * 2u)

typedef __attribute__((ext_vector_type(16))) __bf16 v16bf;
typedef __attribute__((ext_vector_type(8)))  float  v8f;

union Frag { uint4 u[2]; v16bf v; };

__device__ inline unsigned short f2bf(float f) {
    unsigned int u = __float_as_uint(f);
    u += 0x7FFFu + ((u >> 16) & 1u);      // round-to-nearest-even
    return (unsigned short)(u >> 16);
}

// ---- gfx1250 async global->LDS copy (ASYNCcnt-tracked) ----
__device__ inline void async_copy_b128(unsigned lds_byte_off, const void* gaddr) {
    asm volatile("global_load_async_to_lds_b128 %0, %1, off"
                 :: "v"(lds_byte_off), "v"(gaddr) : "memory");
}
template <int N>
__device__ inline void wait_async_le() {
    asm volatile("s_wait_asynccnt %0" :: "i"(N) : "memory");
}

// Issue async copies for `rows` weight rows (Kpad halfwords each, chunks of 16B) into LDS half-buffer.
// rows*kchunks is always a multiple of THREADS -> every thread issues exactly the same count.
__device__ inline void stage_half_async(const unsigned short* __restrict__ src,
                                        unsigned short* dstbuf, int rows, int kchunks, int tid) {
    int total = rows * kchunks;
    int kpad = kchunks << 3;
    for (int idx = tid; idx < total; idx += THREADS) {
        int r = idx / kchunks;
        int c = idx - r * kchunks;
        unsigned lds = (unsigned)(size_t)(dstbuf + r * SW + (c << 3));
        async_copy_b128(lds, (const void*)(src + r * kpad + (c << 3)));
    }
}

// Fallback: synchronous transpose+convert staging straight from fp32 weights.
__device__ inline void stage_half_sync(const float* __restrict__ W, int K, int N,
                                       unsigned short* dstbuf, int rows, int Kpad,
                                       int nbase, int tid) {
    int total = rows * Kpad;
    for (int i = tid; i < total; i += THREADS) {
        int r = i / Kpad;
        int k = i - r * Kpad;
        int n = nbase + r;
        float v = (n < N && k < K) ? W[k * N + n] : 0.0f;
        dstbuf[r * SW + k] = f2bf(v);
    }
}

// A operand: 16x32 bf16 tile. Lanes 0-15: K 0..7 & 16..23; lanes 16-31: K 8..15 & 24..31.
__device__ inline void load_afrags(const unsigned short* act, int mrow, int lane,
                                   int ktiles, v16bf* afr) {
    const unsigned short* p = act + (mrow + (lane & 15)) * SA + ((lane >> 4) << 3);
    for (int t = 0; t < ktiles; ++t) {
        Frag f;
        f.u[0] = *(const uint4*)(p + t * 32);
        f.u[1] = *(const uint4*)(p + t * 32 + 16);
        afr[t] = f.v;
    }
}

// B operand from transposed weights (buffer-relative row base).
__device__ inline v16bf load_bfrag(const unsigned short* wb, int nbase, int kt, int lane) {
    const unsigned short* p = wb + (nbase + (lane & 15)) * SW + kt * 32 + ((lane >> 4) << 3);
    Frag f;
    f.u[0] = *(const uint4*)p;
    f.u[1] = *(const uint4*)(p + 16);
    return f.v;
}

__device__ inline v8f tile_acc(const v16bf* afr, const unsigned short* wb,
                               int nbase, int ktiles, int lane, float bv) {
    v8f acc;
#pragma unroll
    for (int g = 0; g < 8; ++g) acc[g] = bv;
    for (int kt = 0; kt < ktiles; ++kt) {
        v16bf b = load_bfrag(wb, nbase, kt, lane);
        acc = __builtin_amdgcn_wmma_f32_16x16x32_bf16(false, afr[kt], false, b,
                                                      (short)0, acc, false, false);
    }
    return acc;
}

// Compute output tiles [t0, t1) of this wave's 16-row stripe from one weight half-buffer.
__device__ inline void dense_part(const v16bf* afr, const unsigned short* wb,
                                  const float* __restrict__ bias, int ktiles,
                                  int t0, int t1, int nofs, int nvalid,
                                  bool do_relu, int colshift, float* dcol,
                                  unsigned short* actOut, int lane, int mrow) {
    int nl = lane & 15;
    for (int t = t0; t < t1; ++t) {
        int n = t * 16 + nl;
        float bv = (n < nvalid) ? bias[n] : 0.0f;
        v8f acc = tile_acc(afr, wb, t * 16 - nofs, ktiles, lane, bv);
#pragma unroll
        for (int g = 0; g < 8; ++g) {
            int m = mrow + g + ((lane >> 4) << 3);   // C/D layout: VGPR g -> row g / g+8
            float v = acc[g];
            if (do_relu) v = fmaxf(v, 0.0f);
            if (dcol && n == 0) dcol[m * 4 + 3] = v;
            int nc = n + colshift;
            if (n < nvalid && nc >= 0) actOut[m * SA + nc] = f2bf(v);
        }
    }
}

// Full layer: double-buffered async weight staging (half0 -> bufA, half1 -> bufB),
// compute on half0 overlaps half1's in-flight copies. C0 = async loads/thread per half.
template <int C0>
__device__ inline void run_layer(const unsigned short* __restrict__ wsl,
                                 const float* __restrict__ W, int K, int N, int Kpad,
                                 int Nh, int halfTiles, int ntiles, int nvalid,
                                 const float* __restrict__ bias, bool do_relu,
                                 int colshift, float* dcol,
                                 const unsigned short* actIn, unsigned short* actOut,
                                 unsigned short* bufA, unsigned short* bufB,
                                 int use_ws, int tid, int lane, int mrow, int ktiles) {
    int Npad  = ntiles * 16;
    int rows1 = Npad - Nh;
    if (use_ws) {
        stage_half_async(wsl, bufA, Nh, Kpad >> 3, tid);
        stage_half_async(wsl + Nh * Kpad, bufB, rows1, Kpad >> 3, tid);
    } else {
        stage_half_sync(W, K, N, bufA, Nh, Kpad, 0, tid);
        stage_half_sync(W, K, N, bufB, rows1, Kpad, Nh, tid);
    }
    wait_async_le<C0>();                   // half0 resident (completions are in-order)
    __syncthreads();
    v16bf afr[8];
    load_afrags(actIn, mrow, lane, ktiles, afr);
    dense_part(afr, bufA, bias, ktiles, 0, halfTiles, 0, nvalid,
               do_relu, colshift, dcol, actOut, lane, mrow);
    wait_async_le<0>();                    // half1 resident
    __syncthreads();
    dense_part(afr, bufB, bias, ktiles, halfTiles, ntiles, Nh, nvalid,
               do_relu, colshift, dcol, actOut, lane, mrow);
    __syncthreads();
}

// ---- Prologue: fp32 W[k][n] -> bf16 transposed padded W^T[n][k] in workspace ----
__global__ void __launch_bounds__(256)
convert_weights(const float* __restrict__ W, unsigned short* __restrict__ dst,
                int K, int N, int Kpad, int Npad) {
    int i = blockIdx.x * 256 + threadIdx.x;
    if (i >= Npad * Kpad) return;
    int n = i / Kpad;
    int k = i - n * Kpad;
    float v = (n < N && k < K) ? W[k * N + n] : 0.0f;
    dst[n * Kpad + k] = f2bf(v);
}

__global__ void __launch_bounds__(THREADS)
fourier_mlp_kernel(const float* __restrict__ mean, const float* __restrict__ cov,
                   const float* __restrict__ Wd_in,  const float* __restrict__ bd_in,
                   const float* __restrict__ Wd_hid, const float* __restrict__ bd_hid,
                   const float* __restrict__ Wd_out, const float* __restrict__ bd_out,
                   const float* __restrict__ Wc_in,  const float* __restrict__ bc_in,
                   const float* __restrict__ Wc_hid, const float* __restrict__ bc_hid,
                   const float* __restrict__ Wc_out, const float* __restrict__ bc_out,
                   const int* __restrict__ decayscale,
                   const unsigned short* __restrict__ ws, int use_ws,
                   float* __restrict__ out, int npts) {
    extern __shared__ char smem[];
    unsigned short* actA = (unsigned short*)smem;          // 128 x SA
    unsigned short* actB = actA + ROWS * SA;               // 128 x SA
    unsigned short* bufA = actB + ROWS * SA;               // 144 x SW (weight half 0)
    unsigned short* bufB = bufA + 144 * SW;                // 144 x SW (weight half 1)
    float* outbuf = (float*)(bufB + 144 * SW);             // 128 x 4 (rgb + d)

    int tid  = threadIdx.x;
    int lane = tid & 31;
    int wave = tid >> 5;
    int mrow = wave << 4;
    int base = blockIdx.x * ROWS;
    float decay = (float)decayscale[0];

    // ---- Fourier IPE embedding -> actA[:, 0..95] (bf16, zero-padded) ----
    if (tid < ROWS) {
        int gp = base + tid;
        float x[3], vv[3];
#pragma unroll
        for (int d = 0; d < 3; ++d) {
            x[d]  = (gp < npts) ? mean[gp * 3 + d]    : 0.0f;
            vv[d] = (gp < npts) ? cov[gp * 9 + d * 4] : 0.0f;   // diagonal of 3x3
        }
        unsigned short* row = actA + tid * SA;
        row[0] = f2bf(x[0]); row[1] = f2bf(x[1]); row[2] = f2bf(x[2]);
        float fr = 1.0f;
        for (int f = 0; f < NFREQ; ++f) {
            float wc = fminf(fmaxf(decay - (float)f, 0.0f), 1.0f);
            float window = 0.5f * (1.0f - __cosf(3.14159265358979f * wc));
#pragma unroll
            for (int d = 0; d < 3; ++d) {
                float xb  = x[d] * fr;
                float att = __expf(-0.5f * vv[d] * fr * fr) * window;
                row[3 + f * 6 + d]     = f2bf(__sinf(xb) * att);
                row[3 + f * 6 + 3 + d] = f2bf(__cosf(xb) * att);
            }
            fr *= 2.0f;
        }
        for (int k = EMB; k < EMBPAD; ++k) row[k] = 0;
    }

    // ---- Density input layer: 75(->96) x 256 ----
    run_layer<6>(ws + WS_WDIN, Wd_in, EMB, 256, EMBPAD, 128, 8, 16, 256,
                 bd_in, true, 0, nullptr, actA, actB, bufA, bufB,
                 use_ws, tid, lane, mrow, 3);

    // ---- 6 hidden density layers: 256 x 256 ----
    unsigned short* ain = actB;
    unsigned short* aot = actA;
    for (int l = 0; l < 6; ++l) {
        run_layer<16>(ws + WS_HID + l * 65536, Wd_hid + l * 65536, 256, 256, 256,
                      128, 8, 16, 256, bd_hid + l * 256, true, 0, nullptr,
                      ain, aot, bufA, bufB, use_ws, tid, lane, mrow, 8);
        unsigned short* t = ain; ain = aot; aot = t;
    }
    // ain == actB holds last hidden activation

    // ---- Density output: 256 x 257 (pad N->288); col 0 = d -> outbuf, cols 1..256 -> features ----
    run_layer<18>(ws + WS_WDOUT, Wd_out, 256, 257, 256, 144, 9, 18, 257,
                  bd_out, false, -1, outbuf, ain, aot, bufA, bufB,
                  use_ws, tid, lane, mrow, 8);

    // ---- Color input layer: 256 x 128 ----
    run_layer<8>(ws + WS_WCIN, Wc_in, 256, 128, 256, 64, 4, 8, 128,
                 bc_in, true, 0, nullptr, actA, actB, bufA, bufB,
                 use_ws, tid, lane, mrow, 8);

    // ---- 2 color hidden layers: 128 x 128 ----
    ain = actB; aot = actA;
    for (int l = 0; l < 2; ++l) {
        run_layer<4>(ws + WS_WCHID + l * 16384, Wc_hid + l * 16384, 128, 128, 128,
                     64, 4, 8, 128, bc_hid + l * 128, true, 0, nullptr,
                     ain, aot, bufA, bufB, use_ws, tid, lane, mrow, 4);
        unsigned short* t = ain; ain = aot; aot = t;
    }
    // ain == actB holds final color hidden activation

    // ---- Color output: 128 x 3 (pad N->16), no ReLU -> outbuf[m*4 + 0..2] ----
    if (use_ws) stage_half_async(ws + WS_WCOUT, bufA, 16, 16, tid);
    else        stage_half_sync(Wc_out, 128, 3, bufA, 16, 128, 0, tid);
    wait_async_le<0>();
    __syncthreads();
    {
        v16bf afr[4];
        load_afrags(ain, mrow, lane, 4, afr);
        int nl = lane & 15;
        float bv = (nl < 3) ? bc_out[nl] : 0.0f;
        v8f acc = tile_acc(afr, bufA, 0, 4, lane, bv);
        if (nl < 3) {
#pragma unroll
            for (int g = 0; g < 8; ++g) {
                int m = mrow + g + ((lane >> 4) << 3);
                outbuf[m * 4 + nl] = acc[g];
            }
        }
    }
    __syncthreads();

    // ---- Coalesced float4 writeback: [rgb, d] ----
    if (tid < ROWS) {
        int gp = base + tid;
        if (gp < npts) {
            float4 v = *(const float4*)(outbuf + tid * 4);
            *(float4*)(out + gp * 4) = v;
        }
    }
}

extern "C" void kernel_launch(void* const* d_in, const int* in_sizes, int n_in,
                              void* d_out, int out_size, void* d_ws, size_t ws_size,
                              hipStream_t stream) {
    const float* mean   = (const float*)d_in[0];
    const float* cov    = (const float*)d_in[1];
    const float* Wd_in  = (const float*)d_in[2];
    const float* bd_in  = (const float*)d_in[3];
    const float* Wd_hid = (const float*)d_in[4];
    const float* bd_hid = (const float*)d_in[5];
    const float* Wd_out = (const float*)d_in[6];
    const float* bd_out = (const float*)d_in[7];
    const float* Wc_in  = (const float*)d_in[8];
    const float* bc_in  = (const float*)d_in[9];
    const float* Wc_hid = (const float*)d_in[10];
    const float* bc_hid = (const float*)d_in[11];
    const float* Wc_out = (const float*)d_in[12];
    const float* bc_out = (const float*)d_in[13];
    const int*   decay  = (const int*)d_in[14];
    float* out = (float*)d_out;

    int npts   = in_sizes[0] / 3;
    int blocks = (npts + ROWS - 1) / ROWS;

    int use_ws = (ws_size >= WS_TOTAL_BYTES) ? 1 : 0;
    unsigned short* ws = (unsigned short*)d_ws;

    if (use_ws) {
        // Convert all weights once: fp32 -> bf16, transposed + padded, tight rows.
        auto conv = [&](const float* W, unsigned ofs, int K, int N, int Kpad, int Npad) {
            int tot = Npad * Kpad;
            convert_weights<<<(tot + 255) / 256, 256, 0, stream>>>(W, ws + ofs, K, N, Kpad, Npad);
        };
        conv(Wd_in, WS_WDIN, EMB, 256, EMBPAD, 256);
        for (int l = 0; l < 6; ++l)
            conv(Wd_hid + l * 65536, WS_HID + l * 65536u, 256, 256, 256, 256);
        conv(Wd_out, WS_WDOUT, 256, 257, 256, 288);
        conv(Wc_in,  WS_WCIN,  256, 128, 256, 128);
        for (int l = 0; l < 2; ++l)
            conv(Wc_hid + l * 16384, WS_WCHID + l * 16384u, 128, 128, 128, 128);
        conv(Wc_out, WS_WCOUT, 128, 3, 128, 16);
    }

    size_t shmem = (size_t)(2 * ROWS * SA + 2 * 144 * SW) * sizeof(unsigned short)
                 + (size_t)ROWS * 4 * sizeof(float);

    hipFuncSetAttribute((const void*)fourier_mlp_kernel,
                        hipFuncAttributeMaxDynamicSharedMemorySize, (int)shmem);

    fourier_mlp_kernel<<<blocks, THREADS, shmem, stream>>>(
        mean, cov, Wd_in, bd_in, Wd_hid, bd_hid, Wd_out, bd_out,
        Wc_in, bc_in, Wc_hid, bc_hid, Wc_out, bc_out, decay, ws, use_ws, out, npts);
}
